// DiscreteProcessor_47794396070421
// MI455X (gfx1250) — matrix operand order repaired and checked by reading, not compiled, not run on hardware
//
#include <hip/hip_runtime.h>
#include <hip/hip_bf16.h>
#include <math.h>

#define NN   8192
#define DEG  32
#define EE   (NN*DEG)
#define H    128

typedef __attribute__((ext_vector_type(16))) __bf16       v16bf;
typedef __attribute__((ext_vector_type(8)))  float        v8f;
typedef __attribute__((ext_vector_type(4)))  unsigned int v4u;

union FragBf { v16bf v; v4u u[2]; };

__device__ __forceinline__ unsigned short f2bf(float f) {
  unsigned u = __float_as_uint(f);
  unsigned r = 0x7fffu + ((u >> 16) & 1u);     // round-to-nearest-even
  return (unsigned short)((u + r) >> 16);
}

__device__ __forceinline__ v8f wmma_bf16(v16bf a, v16bf b, v8f c) {
  return __builtin_amdgcn_wmma_f32_16x16x32_bf16(false, a, false, b, (short)0, c, false, false);
}

// A-fragment (16x32 bf16, rows row0..row0+15 of row-major [.., ld])
// lane = m + 16*h; elements e<8 -> K=k0+8h+e ; e>=8 -> K=k0+16+8h+(e-8)
__device__ __forceinline__ v16bf fragA(const unsigned short* base, int ld, int row0, int k0, int lane) {
  int m = lane & 15, h = lane >> 4;
  const unsigned short* p = base + (size_t)(row0 + m) * ld + k0 + h * 8;
  FragBf f; f.u[0] = *(const v4u*)p; f.u[1] = *(const v4u*)(p + 16); return f.v;
}
// A-fragment with per-row table indirection (rows picked from small embedding tables)
__device__ __forceinline__ v16bf fragA_tab(const unsigned short* tbl, const int* rows, int row0, int k0, int lane) {
  int m = lane & 15, h = lane >> 4;
  const unsigned short* p = tbl + (size_t)rows[row0 + m] * H + k0 + h * 8;
  FragBf f; f.u[0] = *(const v4u*)p; f.u[1] = *(const v4u*)(p + 16); return f.v;
}
// B-fragment (32x16 bf16) from transposed weights WT[n][k], ld = K
// lane = n + 16*h; elements e<8 -> K=k0+8h+e ; e>=8 -> K=k0+16+8h+(e-8)
__device__ __forceinline__ v16bf fragB(const unsigned short* bT, int ld, int n0, int k0, int lane) {
  int nn = lane & 15, h = lane >> 4;
  const unsigned short* p = bT + (size_t)(n0 + nn) * ld + k0 + h * 8;
  FragBf f; f.u[0] = *(const v4u*)p; f.u[1] = *(const v4u*)(p + 16); return f.v;
}

// ---------------- prep kernels ----------------

// transpose fp32 [K x NC] -> bf16 [NC x K]
__global__ void k_transpose_bf(const float* __restrict__ src, unsigned short* __restrict__ dst, int K, int NC) {
  int o = blockIdx.x * 256 + threadIdx.x;
  if (o >= K * NC) return;
  int n = o / K, k = o - n * K;
  dst[o] = f2bf(src[(size_t)k * NC + n]);
}

__global__ void k_conv_bf(const float* __restrict__ src, unsigned short* __restrict__ dst, int count) {
  int o = blockIdx.x * 256 + threadIdx.x;
  if (o < count) dst[o] = f2bf(src[o]);
}

// per-node: code, node_fts (f32 + bf16), self-loop scalars
__global__ void k_node_prep(const int* __restrict__ node_states, const float* __restrict__ emb_virtual,
                            const int* __restrict__ src, const float* __restrict__ scalars,
                            float* __restrict__ node_fts, unsigned short* __restrict__ node_fts_bf,
                            int* __restrict__ node_code, float* __restrict__ node_scalars) {
  int n = blockIdx.x, t = threadIdx.x;
  int code = node_states[n * 3 + 0] + node_states[n * 3 + 1] * 2 + node_states[n * 3 + 2] * 4;
  float f = emb_virtual[(size_t)(2 * code) * H + t];
  node_fts[(size_t)n * H + t] = f;
  node_fts_bf[(size_t)n * H + t] = f2bf(f);
  if (t == 0) {
    node_code[n] = code;
    float s = 0.f;
    for (int j = 0; j < DEG; ++j) { int e = n * DEG + j; if (src[e] == n) s += scalars[e]; }
    node_scalars[n] = s;
  }
}

__global__ void k_edge_codes(const int* __restrict__ edge_states, const float* __restrict__ scalars,
                             const int* __restrict__ src, const int* __restrict__ dst,
                             const float* __restrict__ node_scalars,
                             int* __restrict__ ecode, int* __restrict__ scode) {
  int e = blockIdx.x * 256 + threadIdx.x;
  if (e >= EE) return;
  int c = edge_states[e * 4 + 0] + edge_states[e * 4 + 1] * 2 + edge_states[e * 4 + 2] * 4 + edge_states[e * 4 + 3] * 8;
  ecode[e] = c;
  float s = scalars[e];
  float rs = node_scalars[dst[e]];
  float ss = node_scalars[src[e]];
  scode[e] = ((s < rs) ? 1 : 0) + 2 * ((ss + s < rs) ? 1 : 0);
}

// ---------------- node GEMMs: Qpre/Kpre/Vn/G1pre (16 nodes/block, 8 waves) ----------------
__global__ __launch_bounds__(256) void k_node_gemm(
    const unsigned short* __restrict__ A,
    const unsigned short* __restrict__ WqT, const unsigned short* __restrict__ WkT,
    const unsigned short* __restrict__ WvT, const unsigned short* __restrict__ G1T,
    float* __restrict__ Qp, float* __restrict__ Kp, float* __restrict__ Vp, float* __restrict__ Gp) {
  int n0 = blockIdx.x * 16;
  int w = threadIdx.x >> 5, lane = threadIdx.x & 31;
  int wt = w >> 1;
  const unsigned short* WT; float* C;
  if      (wt == 0) { WT = WqT; C = Qp; }
  else if (wt == 1) { WT = WkT; C = Kp; }
  else if (wt == 2) { WT = WvT; C = Vp; }
  else              { WT = G1T; C = Gp; }
  int nbase = (w & 1) * 64;
  for (int ct = 0; ct < 4; ++ct) {
    int nc = nbase + ct * 16;
    v8f acc = {0.f,0.f,0.f,0.f,0.f,0.f,0.f,0.f};
    #pragma unroll
    for (int kc = 0; kc < 4; ++kc) {
      v16bf a = fragA(A, H, n0, kc * 32, lane);
      v16bf b = fragB(WT, H, nc, kc * 32, lane);
      acc = wmma_bf16(a, b, acc);
    }
    int nn = lane & 15, g = lane >> 4;
    #pragma unroll
    for (int r = 0; r < 8; ++r) C[(size_t)(n0 + 8 * g + r) * H + nc + nn] = acc[r];
  }
}

// ---------------- node epilogue: LN(Q), LN(K), gate u ----------------
__device__ __forceinline__ float bsum128(float v, float* red) {
  int t = threadIdx.x;
  __syncthreads();
  red[t] = v; __syncthreads();
  for (int s = 64; s > 0; s >>= 1) { if (t < s) red[t] += red[t + s]; __syncthreads(); }
  return red[0];
}

__global__ __launch_bounds__(128) void k_node_epi(
    float* __restrict__ Qp, float* __restrict__ Kp, const float* __restrict__ Gp,
    const float* __restrict__ lnqs, const float* __restrict__ lnqb,
    const float* __restrict__ lnks, const float* __restrict__ lnkb,
    const float* __restrict__ gb1, const float* __restrict__ gW2, const float* __restrict__ gb2,
    float* __restrict__ u) {
  __shared__ float red[128];
  int n = blockIdx.x, t = threadIdx.x;
  // LN(Q)
  float q = Qp[(size_t)n * H + t];
  float mu = bsum128(q, red) * (1.f / H);
  float d = q - mu;
  float var = bsum128(d * d, red) * (1.f / H);
  Qp[(size_t)n * H + t] = d * rsqrtf(var + 1e-5f) * lnqs[t] + lnqb[t];
  // LN(K)
  float k = Kp[(size_t)n * H + t];
  mu = bsum128(k, red) * (1.f / H);
  d = k - mu;
  var = bsum128(d * d, red) * (1.f / H);
  Kp[(size_t)n * H + t] = d * rsqrtf(var + 1e-5f) * lnks[t] + lnkb[t];
  // gate
  float g = fmaxf(Gp[(size_t)n * H + t] + gb1[t], 0.f);
  float sg = bsum128(g * gW2[t], red);
  if (t == 0) u[n] = 1.f / (1.f + expf(-(sg + gb2[0])));
}

// ---------------- edge mega-kernel: 1 node / block, 3 WMMA GEMMs + LN + attention ----------------
__global__ __launch_bounds__(256) void k_edge_attn(
    const float* __restrict__ Q, const float* __restrict__ Kn, const float* __restrict__ Vn,
    const float* __restrict__ u, const float* __restrict__ node_fts,
    const int* __restrict__ node_code, const int* __restrict__ ecode, const int* __restrict__ scode,
    const int* __restrict__ src, const int* __restrict__ rev_idx,
    const unsigned short* __restrict__ recvBf, const unsigned short* __restrict__ edgeBf,
    const unsigned short* __restrict__ statBf,
    const unsigned short* __restrict__ WekT, const unsigned short* __restrict__ WcT,
    const unsigned short* __restrict__ WevT,
    const float* __restrict__ emb_edge_f32,
    const float* __restrict__ lnkes, const float* __restrict__ lnkeb,
    const int* __restrict__ tstep,
    float* __restrict__ out1, float* __restrict__ out2) {
  __shared__ int s_src[DEG], s_sel[DEG], s_ec[DEG], s_rc[DEG], s_sc[DEG];
  __shared__ float s_Q[H];
  __shared__ float s_u;
  __shared__ __align__(16) unsigned short s_comb[DEG * H];   // combined, bf16
  __shared__ float s_K[DEG * H];                             // edge_Kpre -> Kc
  __shared__ float s_V[DEG * H];                             // edge_V -> Vc
  __shared__ float s_logit[DEG + 1], s_attn[DEG + 1], s_agg[H];

  int n = blockIdx.x;
  int t = threadIdx.x, w = t >> 5, lane = t & 31;

  if (t < DEG) {
    int e = n * DEG + t;
    int s = src[e];
    s_src[t] = s;
    s_sel[t] = 2 * node_code[s];
    s_ec[t] = ecode[e];
    s_rc[t] = ecode[rev_idx[e]];
    s_sc[t] = scode[e];
  }
  if (t < H) s_Q[t] = Q[(size_t)n * H + t];
  if (t == 0) s_u = u[n];
  __syncthreads();

  int n0 = 16 * w;
  // GEMM1: sel @ WekT -> s_K (pre-LN), K=128
  for (int rt = 0; rt < DEG; rt += 16) {
    v8f acc = {0.f,0.f,0.f,0.f,0.f,0.f,0.f,0.f};
    #pragma unroll
    for (int kc = 0; kc < 4; ++kc) {
      v16bf a = fragA_tab(recvBf, s_sel, rt, kc * 32, lane);
      v16bf b = fragB(WekT, H, n0, kc * 32, lane);
      acc = wmma_bf16(a, b, acc);
    }
    int nn = lane & 15, g = lane >> 4;
    #pragma unroll
    for (int r = 0; r < 8; ++r) s_K[(rt + 8 * g + r) * H + n0 + nn] = acc[r];
  }
  // GEMM2: concat(edge_fts, edge_fts[rev], static) @ WcombT -> s_comb (bf16), K=384
  for (int rt = 0; rt < DEG; rt += 16) {
    v8f acc = {0.f,0.f,0.f,0.f,0.f,0.f,0.f,0.f};
    #pragma unroll
    for (int kc = 0; kc < 12; ++kc) {
      int k0 = kc * 32;
      const unsigned short* tbl; const int* rows; int kl;
      if (k0 < 128)      { tbl = edgeBf; rows = s_ec; kl = k0; }
      else if (k0 < 256) { tbl = edgeBf; rows = s_rc; kl = k0 - 128; }
      else               { tbl = statBf; rows = s_sc; kl = k0 - 256; }
      v16bf a = fragA_tab(tbl, rows, rt, kl, lane);
      v16bf b = fragB(WcT, 384, n0, k0, lane);
      acc = wmma_bf16(a, b, acc);
    }
    int nn = lane & 15, g = lane >> 4;
    #pragma unroll
    for (int r = 0; r < 8; ++r) s_comb[(rt + 8 * g + r) * H + n0 + nn] = f2bf(acc[r]);
  }
  __syncthreads();

  // GEMM3: combined @ WevT -> s_V, K=128 (A from LDS)
  for (int rt = 0; rt < DEG; rt += 16) {
    v8f acc = {0.f,0.f,0.f,0.f,0.f,0.f,0.f,0.f};
    #pragma unroll
    for (int kc = 0; kc < 4; ++kc) {
      v16bf a = fragA((const unsigned short*)s_comb, H, rt, kc * 32, lane);
      v16bf b = fragB(WevT, H, n0, kc * 32, lane);
      acc = wmma_bf16(a, b, acc);
    }
    int nn = lane & 15, g = lane >> 4;
    #pragma unroll
    for (int r = 0; r < 8; ++r) s_V[(rt + 8 * g + r) * H + n0 + nn] = acc[r];
  }
  // LN over edge_K rows (4 rows per wave), then add Kn[src] -> Kc
  {
    int r0 = w * 4;
    for (int r = r0; r < r0 + 4; ++r) {
      float vv[4];
      #pragma unroll
      for (int i = 0; i < 4; ++i) vv[i] = s_K[r * H + lane * 4 + i];
      float sum = vv[0] + vv[1] + vv[2] + vv[3];
      float sq  = vv[0]*vv[0] + vv[1]*vv[1] + vv[2]*vv[2] + vv[3]*vv[3];
      for (int o = 16; o > 0; o >>= 1) { sum += __shfl_xor(sum, o, 32); sq += __shfl_xor(sq, o, 32); }
      float mu = sum * (1.f / H);
      float var = sq * (1.f / H) - mu * mu;
      float rstd = rsqrtf(var + 1e-5f);
      const float* knrow = Kn + (size_t)s_src[r] * H;
      #pragma unroll
      for (int i = 0; i < 4; ++i) {
        int hh = lane * 4 + i;
        s_K[r * H + hh] = (vv[i] - mu) * rstd * lnkes[hh] + lnkeb[hh] + knrow[hh];
      }
    }
  }
  __syncthreads();

  // Vc = edge_V + Vn[src]; logits
  for (int idx = t; idx < DEG * H; idx += 256) {
    int m = idx >> 7, h = idx & 127;
    s_V[idx] += Vn[(size_t)s_src[m] * H + h];
  }
  if (t < DEG + 1) {
    float s = 0.f;
    if (t == 0) {
      const float* kr = Kn + (size_t)n * H;
      for (int h = 0; h < H; ++h) s += s_Q[h] * kr[h];
    } else {
      const float* kr = &s_K[(t - 1) * H];
      for (int h = 0; h < H; ++h) s += s_Q[h] * kr[h];
    }
    s_logit[t] = s * 0.08838834764831845f;   // 1/sqrt(128)
  }
  __syncthreads();

  // scalar attention pipeline (all 33 slots valid: mask is all-true by construction)
  if (t == 0) {
    const int KK = DEG + 1;
    float lg[DEG + 1], z[DEG + 1], cz[DEG + 1], cz2[DEG + 1], soft[DEG + 1], p[DEG + 1];
    for (int i = 0; i < KK; ++i) { lg[i] = s_logit[i]; z[i] = lg[i]; }
    for (int i = 1; i < KK; ++i) {            // insertion sort, descending
      float key = z[i]; int j = i - 1;
      while (j >= 0 && z[j] < key) { z[j + 1] = z[j]; --j; }
      z[j + 1] = key;
    }
    float a = 0.f, b = 0.f;
    for (int i = 0; i < KK; ++i) { a += z[i]; b += z[i] * z[i]; cz[i] = a; cz2[i] = b; }
    // entmax15
    int k15 = 0; float tauc[DEG + 1];
    for (int k = 1; k <= KK; ++k) {
      float fk = (float)k, mz = cz[k - 1] / fk, mz2 = cz2[k - 1] / fk;
      float dd = fmaxf(mz * mz - mz2 + 1.f / fk, 0.f);
      tauc[k - 1] = mz - sqrtf(dd + 1e-8f);
      if (z[k - 1] > tauc[k - 1]) ++k15;
    }
    float tau15 = tauc[k15 - 1];
    // sparsemax
    int ks = 0;
    for (int k = 1; k <= KK; ++k) if ((float)k * z[k - 1] > cz[k - 1] - 1.f) ++ks;
    float taus = (cz[ks - 1] - 1.f) / (float)ks;
    // softmax
    float mx = lg[0];
    for (int i = 1; i < KK; ++i) mx = fmaxf(mx, lg[i]);
    float se = 0.f;
    for (int i = 0; i < KK; ++i) { soft[i] = expf(lg[i] - mx); se += soft[i]; }
    for (int i = 0; i < KK; ++i) soft[i] /= se;
    // interpolate
    float uu = s_u;
    for (int i = 0; i < KK; ++i) {
      float p15 = fmaxf(lg[i] - tau15, 0.f); p15 *= p15;
      float ps  = fmaxf(lg[i] - taus, 0.f);
      if (uu <= 0.5f) { float wl = uu * 2.f;          p[i] = (1.f - wl) * soft[i] + wl * p15; }
      else            { float wh = (uu - 0.5f) * 2.f; p[i] = (1.f - wh) * p15   + wh * ps;  }
    }
    float hs = 0.f;
    for (int i = 0; i < KK; ++i) hs += (p[i] > 1e-4f) ? 1.f : 0.f;
    int step = tstep[0];
    if (step == -1) {
      for (int i = 0; i < KK; ++i) s_attn[i] = ((p[i] > 1e-4f) ? 1.f : 0.f) / (hs + 1e-9f);
    } else {
      float tanl = powf(0.01f, (float)step / 1000.f);
      float beta = fminf(fmaxf(tanl, 0.01f), 1.f);
      for (int i = 0; i < KK; ++i) {
        float hard = ((p[i] > 1e-4f) ? 1.f : 0.f) / (hs + 1e-9f);
        float gp;
        if (beta > 0.01f) { float sp = soft[i] / (1.f + 1e-9f); gp = (1.f - beta) * p[i] + beta * sp; }
        else gp = p[i];
        s_attn[i] = (hard - gp) + gp;   // straight-through forward value
      }
    }
  }
  __syncthreads();

  // aggregate + node output
  if (t < H) {
    float a = s_attn[0] * Vn[(size_t)n * H + t];
    for (int k = 0; k < DEG; ++k) a += s_attn[k + 1] * s_V[k * H + t];
    out1[(size_t)n * H + t] = node_fts[(size_t)n * H + t] + a;
    s_agg[t] = a;
  }
  __syncthreads();

  // edge output: edge_fts + agg[dst]
  for (int idx = t; idx < DEG * H; idx += 256) {
    int m = idx >> 7, h = idx & 127;
    out2[((size_t)n * DEG + m) * H + h] = emb_edge_f32[(size_t)s_ec[m] * H + h] + s_agg[h];
  }
}

// ---------------- workspace layout ----------------
static constexpr size_t SQB      = (size_t)H * H * 2;                // 128x128 bf16
static constexpr size_t OFF_WQT  = 0;
static constexpr size_t OFF_WKT  = OFF_WQT  + SQB;
static constexpr size_t OFF_WVT  = OFF_WKT  + SQB;
static constexpr size_t OFF_G1T  = OFF_WVT  + SQB;
static constexpr size_t OFF_WEKT = OFF_G1T  + SQB;
static constexpr size_t OFF_WEVT = OFF_WEKT + SQB;
static constexpr size_t OFF_WCT  = OFF_WEVT + SQB;                   // 128x384 bf16
static constexpr size_t OFF_RECB = OFF_WCT  + (size_t)H * 384 * 2;
static constexpr size_t OFF_EDGB = OFF_RECB + (size_t)16 * H * 2;
static constexpr size_t OFF_STAB = OFF_EDGB + (size_t)16 * H * 2;
static constexpr size_t OFF_NCOD = OFF_STAB + (size_t)4 * H * 2;
static constexpr size_t OFF_NSCL = OFF_NCOD + (size_t)NN * 4;
static constexpr size_t OFF_NF   = OFF_NSCL + (size_t)NN * 4;
static constexpr size_t OFF_NFB  = OFF_NF   + (size_t)NN * H * 4;
static constexpr size_t OFF_Q    = OFF_NFB  + (size_t)NN * H * 2;
static constexpr size_t OFF_K    = OFF_Q    + (size_t)NN * H * 4;
static constexpr size_t OFF_V    = OFF_K    + (size_t)NN * H * 4;
static constexpr size_t OFF_G    = OFF_V    + (size_t)NN * H * 4;
static constexpr size_t OFF_U    = OFF_G    + (size_t)NN * H * 4;
static constexpr size_t OFF_EC   = OFF_U    + (size_t)NN * 4;
static constexpr size_t OFF_SC   = OFF_EC   + (size_t)EE * 4;

extern "C" void kernel_launch(void* const* d_in, const int* in_sizes, int n_in,
                              void* d_out, int out_size, void* d_ws, size_t ws_size,
                              hipStream_t stream) {
  const int*   node_states = (const int*)  d_in[0];
  const int*   edge_states = (const int*)  d_in[1];
  const float* scalars     = (const float*)d_in[2];
  const int*   edge_index  = (const int*)  d_in[3];
  const int*   rev_idx     = (const int*)  d_in[4];
  const int*   tstep       = (const int*)  d_in[5];
  const float* emb_virtual = (const float*)d_in[6];
  const float* emb_recv    = (const float*)d_in[7];
  const float* emb_edge    = (const float*)d_in[8];
  const float* emb_static  = (const float*)d_in[9];
  const float* Wq   = (const float*)d_in[10];
  const float* Wk   = (const float*)d_in[11];
  const float* Wv   = (const float*)d_in[12];
  const float* Wek  = (const float*)d_in[13];
  const float* Wev  = (const float*)d_in[14];
  const float* Wcomb= (const float*)d_in[15];
  const float* gW1  = (const float*)d_in[16];
  const float* gb1  = (const float*)d_in[17];
  const float* gW2  = (const float*)d_in[18];
  const float* gb2  = (const float*)d_in[19];
  const float* lnqs = (const float*)d_in[20];
  const float* lnqb = (const float*)d_in[21];
  const float* lnks = (const float*)d_in[22];
  const float* lnkb = (const float*)d_in[23];
  const float* lnkes= (const float*)d_in[24];
  const float* lnkeb= (const float*)d_in[25];

  const int* src = edge_index;
  const int* dst = edge_index + EE;

  unsigned char* ws = (unsigned char*)d_ws;
  unsigned short* WqT  = (unsigned short*)(ws + OFF_WQT);
  unsigned short* WkT  = (unsigned short*)(ws + OFF_WKT);
  unsigned short* WvT  = (unsigned short*)(ws + OFF_WVT);
  unsigned short* G1T  = (unsigned short*)(ws + OFF_G1T);
  unsigned short* WekT = (unsigned short*)(ws + OFF_WEKT);
  unsigned short* WevT = (unsigned short*)(ws + OFF_WEVT);
  unsigned short* WcT  = (unsigned short*)(ws + OFF_WCT);
  unsigned short* recB = (unsigned short*)(ws + OFF_RECB);
  unsigned short* edgB = (unsigned short*)(ws + OFF_EDGB);
  unsigned short* staB = (unsigned short*)(ws + OFF_STAB);
  int*   ncode = (int*)  (ws + OFF_NCOD);
  float* nscal = (float*)(ws + OFF_NSCL);
  float* nfts  = (float*)(ws + OFF_NF);
  unsigned short* nftsB = (unsigned short*)(ws + OFF_NFB);
  float* Qb = (float*)(ws + OFF_Q);
  float* Kb = (float*)(ws + OFF_K);
  float* Vb = (float*)(ws + OFF_V);
  float* Gb = (float*)(ws + OFF_G);
  float* ub = (float*)(ws + OFF_U);
  int* ecode = (int*)(ws + OFF_EC);
  int* scode = (int*)(ws + OFF_SC);

  float* out1 = (float*)d_out;
  float* out2 = out1 + (size_t)NN * H;

  // weight transposes + table conversions
  int sq = H * H, bc = (sq + 255) / 256;
  k_transpose_bf<<<bc, 256, 0, stream>>>(Wq,  WqT,  H, H);
  k_transpose_bf<<<bc, 256, 0, stream>>>(Wk,  WkT,  H, H);
  k_transpose_bf<<<bc, 256, 0, stream>>>(Wv,  WvT,  H, H);
  k_transpose_bf<<<bc, 256, 0, stream>>>(gW1, G1T,  H, H);
  k_transpose_bf<<<bc, 256, 0, stream>>>(Wek, WekT, H, H);
  k_transpose_bf<<<bc, 256, 0, stream>>>(Wev, WevT, H, H);
  k_transpose_bf<<<(384 * H + 255) / 256, 256, 0, stream>>>(Wcomb, WcT, 384, H);
  k_conv_bf<<<(16 * H + 255) / 256, 256, 0, stream>>>(emb_recv,   recB, 16 * H);
  k_conv_bf<<<(16 * H + 255) / 256, 256, 0, stream>>>(emb_edge,   edgB, 16 * H);
  k_conv_bf<<<(4  * H + 255) / 256, 256, 0, stream>>>(emb_static, staB, 4 * H);

  // node prep + edge codes
  k_node_prep<<<NN, 128, 0, stream>>>(node_states, emb_virtual, src, scalars, nfts, nftsB, ncode, nscal);
  k_edge_codes<<<EE / 256, 256, 0, stream>>>(edge_states, scalars, src, dst, nscal, ecode, scode);

  // node GEMMs + epilogue
  k_node_gemm<<<NN / 16, 256, 0, stream>>>(nftsB, WqT, WkT, WvT, G1T, Qb, Kb, Vb, Gb);
  k_node_epi<<<NN, 128, 0, stream>>>(Qb, Kb, Gb, lnqs, lnqb, lnks, lnkb, gb1, gW2, gb2, ub);

  // fused per-node edge GEMMs + attention + outputs
  k_edge_attn<<<NN, 256, 0, stream>>>(Qb, Kb, Vb, ub, nfts, ncode, ecode, scode, src, rev_idx,
                                      recB, edgB, staB, WekT, WcT, WevT,
                                      emb_edge, lnkes, lnkeb, tstep, out1, out2);
}
